// Pure3DUnet_51565377356544
// MI455X (gfx1250) — compile-verified, matrix operand order checked by reading, not run on hardware
//
#include <hip/hip_runtime.h>

// ---------------------------------------------------------------------------
// Sizes (fixed by the reference)
// ---------------------------------------------------------------------------
#define B_    2
#define P_    16384
#define CIMG  256
#define V_    2048
#define C_    512
#define NH    8
#define HD    64          // C_/NH
#define NG    32
#define GW    16          // C_/NG

typedef __attribute__((ext_vector_type(16))) _Float16 v16h;
typedef __attribute__((ext_vector_type(8)))  _Float16 v8h;
typedef __attribute__((ext_vector_type(4)))  _Float16 v4h;
typedef __attribute__((ext_vector_type(8)))  float    v8f;
typedef __attribute__((ext_vector_type(4)))  float    v4f;

#define WMMA_F16(a, b, c) \
  __builtin_amdgcn_wmma_f32_16x16x32_f16(false, (a), false, (b), (short)0, (c), false, false)

// --- helpers ----------------------------------------------------------------
static __device__ __forceinline__ v8h cvt8(const float* p) {
  v4f a = *(const v4f*)p;
  v4f b = *(const v4f*)(p + 4);
  v8h r;
  r[0] = (_Float16)a[0]; r[1] = (_Float16)a[1]; r[2] = (_Float16)a[2]; r[3] = (_Float16)a[3];
  r[4] = (_Float16)b[0]; r[5] = (_Float16)b[1]; r[6] = (_Float16)b[2]; r[7] = (_Float16)b[3];
  return r;
}
static __device__ __forceinline__ v4h cvt4(v4f a) {
  v4h r;
  r[0] = (_Float16)a[0]; r[1] = (_Float16)a[1]; r[2] = (_Float16)a[2]; r[3] = (_Float16)a[3];
  return r;
}
// 16-bit A/B fragment whose K-run is contiguous: elements 0..7 <- p[0..7],
// elements 8..15 <- p[16..23]   (p must already include the +kb lane offset)
static __device__ __forceinline__ v16h load_frag_f32(const float* p) {
  v8h lo = cvt8(p), hi = cvt8(p + 16);
  return __builtin_shufflevector(lo, hi, 0,1,2,3,4,5,6,7,8,9,10,11,12,13,14,15);
}
static __device__ __forceinline__ v16h load_frag_lds(const _Float16* p) {
  v8h lo = *(const v8h*)p;          // ds_load_b128
  v8h hi = *(const v8h*)(p + 16);   // ds_load_b128
  return __builtin_shufflevector(lo, hi, 0,1,2,3,4,5,6,7,8,9,10,11,12,13,14,15);
}

// ---------------------------------------------------------------------------
// 1) zero workspace region (scatter accumulators) -- required every call
// ---------------------------------------------------------------------------
__global__ void zero_f32(float* __restrict__ p, int n) {
  int i = blockIdx.x * blockDim.x + threadIdx.x;
  if (i < n) p[i] = 0.0f;
}

// ---------------------------------------------------------------------------
// 2) masked scatter-add of pixel features into voxels (atomics; L2 resident)
// ---------------------------------------------------------------------------
__global__ __launch_bounds__(CIMG)
void scatter_accum(const float* __restrict__ feats, const int* __restrict__ vid,
                   const int* __restrict__ mask, float* __restrict__ sums,
                   float* __restrict__ cnts) {
  int pix = blockIdx.x;                  // 0 .. B*P-1
  int c   = threadIdx.x;                 // 0 .. 255
  if (mask[pix] == 0) return;
  int b   = pix >> 14;                   // P_ = 16384
  int seg = b * V_ + vid[pix];
  atomicAdd(&sums[(size_t)seg * CIMG + c], feats[(size_t)pix * CIMG + c]);
  if (c == 0) atomicAdd(&cnts[seg], 1.0f);
}

// ---------------------------------------------------------------------------
// 3) vox = sums / max(cnt,1)   (in place)
// ---------------------------------------------------------------------------
__global__ void voxel_mean(float* __restrict__ sums, const float* __restrict__ cnts, int n) {
  int i = blockIdx.x * blockDim.x + threadIdx.x;
  if (i < n) sums[i] = sums[i] / fmaxf(cnts[i / CIMG], 1.0f);
}

// ---------------------------------------------------------------------------
// 4/7/9) WMMA GEMM:  out[M,N] = A[M,K] @ W[K,N] (+bias) (+residual)
//   Block = 8 waves = 16x128 tile. Per 32-K slab:
//     * cooperative coalesced b128 loads; A staged natural, W staged TRANSPOSED
//       ([col][k], stride 40 halves) so both fragments are 2x ds_load_b128.
// ---------------------------------------------------------------------------
#define WT_STRIDE 40                       // 80B rows: 16B aligned, bank-staggered
__global__ __launch_bounds__(256)
void gemm_wmma_f16(const float* __restrict__ A, const float* __restrict__ W,
                   const float* __restrict__ bias, const float* __restrict__ resid,
                   float* __restrict__ out, int M, int N, int K) {
  __shared__ _Float16 A2[16 * 32];            // 1 KB,  [row][k]
  __shared__ _Float16 W2t[128 * WT_STRIDE];   // 10 KB, [col][k] transposed
  int nbN  = N >> 7;                          // N / 128 block-columns
  int tM   = (blockIdx.x / nbN) << 4;
  int tNb  = (blockIdx.x % nbN) << 7;
  int tid  = threadIdx.x;
  int wib  = tid >> 5;                        // wave in block: owns cols wib*16..
  int lane = tid & 31;
  int r    = lane & 15;
  int half = lane >> 4;
  int kb   = half * 8;

  v8f acc = {};
  for (int k0 = 0; k0 < K; k0 += 32) {
    __syncthreads();                          // protect LDS reuse across slabs
    // --- stage A tile: 16 rows x 32 k, threads 0..127, one float4 each ---
    if (tid < 128) {
      int row = tid >> 3, c4 = (tid & 7) * 4;
      v4f v = *(const v4f*)&A[(size_t)(tM + row) * K + k0 + c4];
      *(v4h*)&A2[row * 32 + c4] = cvt4(v);
    }
    // --- branchless prefetch of next W slab (one per thread) ---
    {
      int kp = (k0 + 32 < K) ? (k0 + 32) : k0;
      __builtin_prefetch(&W[(size_t)(kp + (tid >> 5)) * N + tNb + (tid & 31) * 4], 0, 3);
    }
    // --- stage W tile 32k x 128col, transposed into LDS ---
#pragma unroll
    for (int i = 0; i < 4; ++i) {
      int idx = i * 256 + tid;                // 0..1023
      int row = idx >> 5, c4 = (idx & 31) * 4;
      v4h h = cvt4(*(const v4f*)&W[(size_t)(k0 + row) * N + tNb + c4]);
      W2t[(c4 + 0) * WT_STRIDE + row] = h[0];
      W2t[(c4 + 1) * WT_STRIDE + row] = h[1];
      W2t[(c4 + 2) * WT_STRIDE + row] = h[2];
      W2t[(c4 + 3) * WT_STRIDE + row] = h[3];
    }
    __syncthreads();
    // --- fragments (all 2x ds_load_b128) + WMMA ---
    v16h a  = load_frag_lds(&A2[r * 32 + kb]);
    v16h bf = load_frag_lds(&W2t[(wib * 16 + r) * WT_STRIDE + kb]);
    acc = WMMA_F16(a, bf, acc);
  }
#pragma unroll
  for (int j = 0; j < 8; ++j) {
    int row = tM + j + half * 8;
    int col = tNb + wib * 16 + r;
    float v = acc[j];
    if (bias)  v += bias[col];
    if (resid) v += resid[(size_t)row * N + col];
    out[(size_t)row * N + col] = v;
  }
}

// ---------------------------------------------------------------------------
// 5) GroupNorm stats: one block per (b,g); reduce over V_*GW = 32768 elements
// ---------------------------------------------------------------------------
__global__ __launch_bounds__(256)
void gn_stats(const float* __restrict__ x, float* __restrict__ stats) {
  __shared__ float ssum[256], ssq[256];
  int bg = blockIdx.x;                    // 0 .. B_*NG-1
  int b = bg / NG, g = bg % NG;
  const int n = V_ * GW;
  float s = 0.0f, q = 0.0f;
  for (int i = threadIdx.x; i < n; i += 256) {
    int vv = i >> 4, c = i & 15;
    float v = x[((size_t)b * V_ + vv) * C_ + g * GW + c];
    s += v; q += v * v;
  }
  ssum[threadIdx.x] = s; ssq[threadIdx.x] = q;
  __syncthreads();
  for (int off = 128; off > 0; off >>= 1) {
    if (threadIdx.x < off) {
      ssum[threadIdx.x] += ssum[threadIdx.x + off];
      ssq[threadIdx.x]  += ssq[threadIdx.x + off];
    }
    __syncthreads();
  }
  if (threadIdx.x == 0) {
    float mean = ssum[0] / (float)n;
    float var  = ssq[0] / (float)n - mean * mean;
    stats[bg * 2 + 0] = mean;
    stats[bg * 2 + 1] = rsqrtf(var + 1e-5f);
  }
}

// ---------------------------------------------------------------------------
// 6) GroupNorm apply: gn = (x - mu) * rstd * gamma + beta
// ---------------------------------------------------------------------------
__global__ void gn_apply(const float* __restrict__ x, const float* __restrict__ stats,
                         const float* __restrict__ gamma, const float* __restrict__ beta,
                         float* __restrict__ gn) {
  int i  = blockIdx.x * blockDim.x + threadIdx.x;   // over B_*V_*C_
  int c  = i & (C_ - 1);
  int bv = i / C_;
  int b  = bv >> 11;                                // V_ = 2048
  int g  = c >> 4;                                  // GW = 16
  float mean = stats[(b * NG + g) * 2 + 0];
  float rstd = stats[(b * NG + g) * 2 + 1];
  gn[i] = (x[i] - mean) * rstd * gamma[c] + beta[c];
}

// ---------------------------------------------------------------------------
// 8) flash attention: one wave per (b,h,16-query tile); 32-key inner steps.
//    Q/K^T fragments: b128 vector loads (d contiguous).
//    V tile staged TRANSPOSED into LDS ([d][kk], stride 40) so each P.V
//    B-fragment is 2x ds_load_b128. Probs round-trip LDS likewise.
// ---------------------------------------------------------------------------
#define VT_STRIDE 40
#define AT_WAVE_LDS (512 + 64 * VT_STRIDE)   // probs 16x32 + VtT 64x40 (f16)
__global__ __launch_bounds__(256)
void flash_attn(const float* __restrict__ qkv, float* __restrict__ out) {
  __shared__ _Float16 sw[8 * AT_WAVE_LDS];          // 48 KB / block
  int wave = (blockIdx.x * blockDim.x + threadIdx.x) >> 5;
  if (wave >= B_ * NH * (V_ / 16)) return;          // wave-uniform
  int wib  = threadIdx.x >> 5;
  int qt   = wave & 127;                            // V_/16 = 128
  int bh   = wave >> 7;
  int b    = bh >> 3, h = bh & 7;
  int lane = threadIdx.x & 31;
  int r    = lane & 15;
  int half = lane >> 4;
  int kb   = half * 8;

  const float* base = qkv + (size_t)b * V_ * (3 * C_);
  const int hoff = h * (3 * HD);                    // 192
  _Float16* pw  = &sw[wib * AT_WAVE_LDS];           // probs [16][32]
  _Float16* VtT = pw + 512;                         // V^T tile [64 d][40]

  // --- load Q (16x64) once: two K=32 fragments, vectorized ---
  const float* qptr = base + (size_t)(qt * 16 + r) * (3 * C_) + hoff;
  v16h q0 = load_frag_f32(qptr + kb);
  v16h q1 = load_frag_f32(qptr + 32 + kb);

  v8f o0 = {}, o1 = {}, o2 = {}, o3 = {};
  float mrow[8], lrow[8];
#pragma unroll
  for (int j = 0; j < 8; ++j) { mrow[j] = -1e30f; lrow[j] = 0.0f; }
  const float scale = 0.125f;                       // 1/sqrt(64)

  for (int k0 = 0; k0 < V_; k0 += 32) {
    // --- K^T fragments: two key tiles x two d-chunks (d contiguous) ---
    const float* kp0 = base + (size_t)(k0 + r)      * (3 * C_) + hoff + HD;
    const float* kp1 = base + (size_t)(k0 + 16 + r) * (3 * C_) + hoff + HD;
    v16h kt00 = load_frag_f32(kp0 + kb);
    v16h kt01 = load_frag_f32(kp0 + 32 + kb);
    v16h kt10 = load_frag_f32(kp1 + kb);
    v16h kt11 = load_frag_f32(kp1 + 32 + kb);

    // --- stage V tile transposed: lane = key row kk, write [d][kk] ---
    {
      const float* vp = base + (size_t)(k0 + lane) * (3 * C_) + hoff + 2 * HD;
#pragma unroll
      for (int c4 = 0; c4 < 64; c4 += 4) {
        v4h hh = cvt4(*(const v4f*)&vp[c4]);        // coalesced b128 load
        VtT[(c4 + 0) * VT_STRIDE + lane] = hh[0];   // b16 stores (overlap WMMA)
        VtT[(c4 + 1) * VT_STRIDE + lane] = hh[1];
        VtT[(c4 + 2) * VT_STRIDE + lane] = hh[2];
        VtT[(c4 + 3) * VT_STRIDE + lane] = hh[3];
      }
    }

    v8f s0 = {}, s1 = {};
    s0 = WMMA_F16(q0, kt00, s0);  s0 = WMMA_F16(q1, kt01, s0);
    s1 = WMMA_F16(q0, kt10, s1);  s1 = WMMA_F16(q1, kt11, s1);

    // --- online softmax over the 32 new key columns ---
    float corr[8];
#pragma unroll
    for (int j = 0; j < 8; ++j) {
      float a0 = s0[j] * scale, a1 = s1[j] * scale;
      float mj = fmaxf(a0, a1);
#pragma unroll
      for (int off = 8; off > 0; off >>= 1)
        mj = fmaxf(mj, __shfl_xor(mj, off, 16));    // row max within 16-lane half
      float mnew = fmaxf(mrow[j], mj);
      float p0 = __expf(a0 - mnew), p1 = __expf(a1 - mnew);
      float ps = p0 + p1;
#pragma unroll
      for (int off = 8; off > 0; off >>= 1)
        ps += __shfl_xor(ps, off, 16);              // row sum within 16-lane half
      corr[j] = __expf(mrow[j] - mnew);
      lrow[j] = lrow[j] * corr[j] + ps;
      mrow[j] = mnew;
      int m = j + half * 8;                         // C/D layout row
      pw[m * 32 + r]      = (_Float16)p0;           // probs -> LDS (A layout next)
      pw[m * 32 + 16 + r] = (_Float16)p1;
    }
#pragma unroll
    for (int j = 0; j < 8; ++j) {
      o0[j] *= corr[j]; o1[j] *= corr[j]; o2[j] *= corr[j]; o3[j] *= corr[j];
    }

    // DS ops are in-order within a wave; fence the compiler + drain DScnt
    asm volatile("s_wait_dscnt 0" ::: "memory");

    // P as A fragment (16x32): 2x ds_load_b128
    v16h pa = load_frag_lds(&pw[r * 32 + kb]);
    // V fragments from transposed LDS: each 2x ds_load_b128
    v16h vb0 = load_frag_lds(&VtT[(0 * 16 + r) * VT_STRIDE + kb]);
    v16h vb1 = load_frag_lds(&VtT[(1 * 16 + r) * VT_STRIDE + kb]);
    v16h vb2 = load_frag_lds(&VtT[(2 * 16 + r) * VT_STRIDE + kb]);
    v16h vb3 = load_frag_lds(&VtT[(3 * 16 + r) * VT_STRIDE + kb]);
    o0 = WMMA_F16(pa, vb0, o0);
    o1 = WMMA_F16(pa, vb1, o1);
    o2 = WMMA_F16(pa, vb2, o2);
    o3 = WMMA_F16(pa, vb3, o3);
  }

  // out[b, row, h*64 + d]  (== transpose(0,2,1,3).reshape)
#pragma unroll
  for (int j = 0; j < 8; ++j) {
    int row = qt * 16 + j + half * 8;
    float inv = 1.0f / lrow[j];
    size_t ob = ((size_t)b * V_ + row) * C_ + h * HD;
    out[ob + r]      = o0[j] * inv;
    out[ob + 16 + r] = o1[j] * inv;
    out[ob + 32 + r] = o2[j] * inv;
    out[ob + 48 + r] = o3[j] * inv;
  }
}

// ---------------------------------------------------------------------------
// Host-side orchestration. Workspace layout (floats), total ~54.5 MB:
//   sums/vox [B*V*CIMG] | cnts [B*V] | x [B*V*C] | gn [B*V*C]
//   | qkv [B*V*3C] | attn [B*V*C] | stats [B*NG*2]
// ---------------------------------------------------------------------------
extern "C" void kernel_launch(void* const* d_in, const int* in_sizes, int n_in,
                              void* d_out, int out_size, void* d_ws, size_t ws_size,
                              hipStream_t stream) {
  (void)in_sizes; (void)n_in; (void)out_size; (void)ws_size;
  const float* pixel_feats = (const float*)d_in[0];
  const float* mix_W  = (const float*)d_in[1];
  const float* mix_b  = (const float*)d_in[2];
  const float* gamma  = (const float*)d_in[3];
  const float* beta   = (const float*)d_in[4];
  const float* qkv_W  = (const float*)d_in[5];
  const float* qkv_b  = (const float*)d_in[6];
  const float* proj_W = (const float*)d_in[7];
  const float* proj_b = (const float*)d_in[8];
  const int*   vid    = (const int*)d_in[9];
  const int*   mask   = (const int*)d_in[10];
  float* out = (float*)d_out;

  float* ws    = (float*)d_ws;
  float* sums  = ws;                                  // also vox (in-place mean)
  float* cnts  = sums + (size_t)B_ * V_ * CIMG;
  float* x     = cnts + (size_t)B_ * V_;
  float* gn    = x    + (size_t)B_ * V_ * C_;
  float* qkv   = gn   + (size_t)B_ * V_ * C_;
  float* attn  = qkv  + (size_t)B_ * V_ * 3 * C_;
  float* stats = attn + (size_t)B_ * V_ * C_;

  // 1) zero scatter accumulators (must happen every call)
  int nz = B_ * V_ * CIMG + B_ * V_;
  zero_f32<<<(nz + 255) / 256, 256, 0, stream>>>(sums, nz);

  // 2) masked scatter-add
  scatter_accum<<<B_ * P_, CIMG, 0, stream>>>(pixel_feats, vid, mask, sums, cnts);

  // 3) divide by counts -> vox
  int nv = B_ * V_ * CIMG;
  voxel_mean<<<nv / 256, 256, 0, stream>>>(sums, cnts, nv);

  // 4) mix GEMM: [4096,256] x [256,512] + bias
  {
    int M = B_ * V_, N = C_, K = CIMG;
    gemm_wmma_f16<<<(M / 16) * (N / 128), 256, 0, stream>>>(sums, mix_W, mix_b, nullptr, x, M, N, K);
  }

  // 5/6) GroupNorm
  gn_stats<<<B_ * NG, 256, 0, stream>>>(x, stats);
  gn_apply<<<(B_ * V_ * C_) / 256, 256, 0, stream>>>(x, stats, gamma, beta, gn);

  // 7) qkv GEMM: [4096,512] x [512,1536] + bias
  {
    int M = B_ * V_, N = 3 * C_, K = C_;
    gemm_wmma_f16<<<(M / 16) * (N / 128), 256, 0, stream>>>(gn, qkv_W, qkv_b, nullptr, qkv, M, N, K);
  }

  // 8) flash attention: 2048 waves
  flash_attn<<<(B_ * NH * (V_ / 16)) / 8, 256, 0, stream>>>(qkv, attn);

  // 9) proj GEMM + bias + residual -> d_out
  {
    int M = B_ * V_, N = C_, K = C_;
    gemm_wmma_f16<<<(M / 16) * (N / 128), 256, 0, stream>>>(attn, proj_W, proj_b, x, out, M, N, K);
  }
}
